// AttentionLoop_54571854463851
// MI455X (gfx1250) — compile-verified, matrix operand order checked by reading, not run on hardware
//
#include <hip/hip_runtime.h>
#include <math.h>

typedef __attribute__((ext_vector_type(4)))  __bf16 v4bf;
typedef __attribute__((ext_vector_type(8)))  __bf16 v8bf;
typedef __attribute__((ext_vector_type(16))) __bf16 v16bf;
typedef __attribute__((ext_vector_type(8)))  float  v8f;

#define S_ 2048
#define B_ 32
#define D_ 1024
#define E_ 1024
#define LDA 1032   // padded LDS row stride (bf16 elems): 2064 B, 16B-aligned chunks

__device__ __forceinline__ float fast_tanh(float x) {
#if __has_builtin(__builtin_amdgcn_tanhf)
  return __builtin_amdgcn_tanhf(x);          // gfx1250 v_tanh_f32
#elif __has_builtin(__builtin_amdgcn_tanh_f32)
  return __builtin_amdgcn_tanh_f32(x);
#else
  return tanhf(x);
#endif
}

// Load one 16-element bf16 fragment: two 16B chunks at +0 and +16 elements,
// per the ISA 16-bit A/B fragment layout (K blocks {0..7,16..23} / {8..15,24..31}).
__device__ __forceinline__ v16bf load_frag(const __bf16* p) {
  const v8bf* vp = (const v8bf*)p;
  v8bf lo = vp[0], hi = vp[2];
  v16bf r;
#pragma unroll
  for (int i = 0; i < 8; ++i) { r[i] = lo[i]; r[i + 8] = hi[i]; }
  return r;
}

#define WMMA_BF16(A, Bf, C) \
  __builtin_amdgcn_wmma_f32_16x16x32_bf16(false, (A), false, (Bf), (short)0, (C), false, false)

// ---------------- Pass 0: W_fc fp32 -> bf16 (one-time, lives in L2) --------
__global__ __launch_bounds__(256) void wconv_kernel(const float* __restrict__ W,
                                                    __bf16* __restrict__ Wb) {
  int idx = (blockIdx.x * 256 + threadIdx.x) * 4;
  float4 f = *(const float4*)(W + idx);
  v4bf o = { (__bf16)f.x, (__bf16)f.y, (__bf16)f.z, (__bf16)f.w };
  *(v4bf*)(Wb + idx) = o;
}

// ---------------- Pass 1: fused (enc+dec) @ Wfc^T -> tanh -> ·w -> scores --
// One workgroup per s. 32 rows (b=0..31) x K=1024 A-tile in LDS (bf16).
// 512 threads = 16 waves; wave w owns e-columns [w*64, w*64+64) as 4 n-tiles.
// B fragments use a ping-pong pair (b0/b1) with a 2x-unrolled K loop: loads
// land directly in the buffer consumed next half-step -> no rotation movs, no
// WMMA->VALU hazard NOPs, one full half-step of load latency always in flight.
__global__ __launch_bounds__(512) void score_kernel(
    const float* __restrict__ enc, const float* __restrict__ dec,
    const __bf16* __restrict__ Wb, const float* __restrict__ sw,
    float* __restrict__ scores) {
  __shared__ __bf16 As[32 * LDA];
  __shared__ float sAcc[32];

  const int tid = threadIdx.x;
  const int s   = blockIdx.x;

  if (tid < 32) sAcc[tid] = 0.0f;

  // Cooperative A-tile load: 32x1024 fp32, add dec row, convert to bf16.
  for (int base = tid * 4; base < 32 * 1024; base += 512 * 4) {
    int row = base >> 10, col = base & 1023;
    float4 e4 = *(const float4*)(enc + ((size_t)(s * 32 + row)) * 1024 + col);
    float4 d4 = *(const float4*)(dec + row * 1024 + col);
    v4bf o = { (__bf16)(e4.x + d4.x), (__bf16)(e4.y + d4.y),
               (__bf16)(e4.z + d4.z), (__bf16)(e4.w + d4.w) };
    *(v4bf*)(&As[row * LDA + col]) = o;
  }
  __syncthreads();

  const int lane   = tid & 31;
  const int wave   = tid >> 5;
  const int col    = lane & 15;     // N column within tile / M row of A frag
  const int half   = lane >> 4;     // which 16-lane half
  const int koff   = half * 8;      // K sub-chunk select per ISA layout
  const int e_base = wave * 64;

  // Per-lane base pointers for the 4 W rows this lane touches.
  const __bf16* wp[4];
#pragma unroll
  for (int nt = 0; nt < 4; ++nt)
    wp[nt] = Wb + (size_t)(e_base + nt * 16 + col) * 1024 + koff;

  v8f acc[4][2];
  v8f zf;
#pragma unroll
  for (int i = 0; i < 8; ++i) zf[i] = 0.0f;
#pragma unroll
  for (int nt = 0; nt < 4; ++nt) { acc[nt][0] = zf; acc[nt][1] = zf; }

  // Prime ping buffer with K-step 0.
  v16bf b0[4], b1[4];
#pragma unroll
  for (int nt = 0; nt < 4; ++nt) b0[nt] = load_frag(wp[nt]);

#pragma unroll 1
  for (int k0 = 0; k0 < 1024 - 64; k0 += 64) {
    // ---- half-step A: consume b0, prefetch b1 (k0+32) ----
    v16bf a0 = load_frag(&As[col * LDA + k0 + koff]);
    v16bf a1 = load_frag(&As[(col + 16) * LDA + k0 + koff]);
#pragma unroll
    for (int nt = 0; nt < 4; ++nt) b1[nt] = load_frag(wp[nt] + k0 + 32);
#pragma unroll
    for (int nt = 0; nt < 4; ++nt) {
      acc[nt][0] = WMMA_BF16(a0, b0[nt], acc[nt][0]);
      acc[nt][1] = WMMA_BF16(a1, b0[nt], acc[nt][1]);
    }
    // ---- half-step B: consume b1, prefetch b0 (k0+64) ----
    v16bf a2 = load_frag(&As[col * LDA + k0 + 32 + koff]);
    v16bf a3 = load_frag(&As[(col + 16) * LDA + k0 + 32 + koff]);
#pragma unroll
    for (int nt = 0; nt < 4; ++nt) b0[nt] = load_frag(wp[nt] + k0 + 64);
#pragma unroll
    for (int nt = 0; nt < 4; ++nt) {
      acc[nt][0] = WMMA_BF16(a2, b1[nt], acc[nt][0]);
      acc[nt][1] = WMMA_BF16(a3, b1[nt], acc[nt][1]);
    }
  }
  {  // Peeled final iteration (k0 = 960): second prefetch would be OOB.
    const int k0 = 1024 - 64;
    v16bf a0 = load_frag(&As[col * LDA + k0 + koff]);
    v16bf a1 = load_frag(&As[(col + 16) * LDA + k0 + koff]);
#pragma unroll
    for (int nt = 0; nt < 4; ++nt) b1[nt] = load_frag(wp[nt] + k0 + 32);
#pragma unroll
    for (int nt = 0; nt < 4; ++nt) {
      acc[nt][0] = WMMA_BF16(a0, b0[nt], acc[nt][0]);
      acc[nt][1] = WMMA_BF16(a1, b0[nt], acc[nt][1]);
    }
    v16bf a2 = load_frag(&As[col * LDA + k0 + 32 + koff]);
    v16bf a3 = load_frag(&As[(col + 16) * LDA + k0 + 32 + koff]);
#pragma unroll
    for (int nt = 0; nt < 4; ++nt) {
      acc[nt][0] = WMMA_BF16(a2, b1[nt], acc[nt][0]);
      acc[nt][1] = WMMA_BF16(a3, b1[nt], acc[nt][1]);
    }
  }

  // tanh -> * score_w[e] -> per-lane partial column sums.
  float c0[8], c1[8];
#pragma unroll
  for (int i = 0; i < 8; ++i) { c0[i] = 0.0f; c1[i] = 0.0f; }
#pragma unroll
  for (int nt = 0; nt < 4; ++nt) {
    float wc = sw[e_base + nt * 16 + col];
#pragma unroll
    for (int i = 0; i < 8; ++i) {
      c0[i] += fast_tanh(acc[nt][0][i]) * wc;
      c1[i] += fast_tanh(acc[nt][1][i]) * wc;
    }
  }
  // Reduce across the 16 lanes that share the same output rows.
#pragma unroll
  for (int m = 8; m >= 1; m >>= 1) {
#pragma unroll
    for (int i = 0; i < 8; ++i) {
      c0[i] += __shfl_xor(c0[i], m, 32);
      c1[i] += __shfl_xor(c1[i], m, 32);
    }
  }
  if (col == 0) {
    int rbase = half * 8;   // lanes 0..15: M=i ; lanes 16..31: M=i+8
#pragma unroll
    for (int i = 0; i < 8; ++i) {
      atomicAdd(&sAcc[rbase + i],      c0[i]);   // m-tile0: b = 0..15
      atomicAdd(&sAcc[16 + rbase + i], c1[i]);   // m-tile1: b = 16..31
    }
  }
  __syncthreads();
  if (tid < 32) scores[(size_t)tid * S_ + s] = sAcc[tid];  // (B,S) layout
}

// ---------------- Pass 2: softmax over S per batch row ---------------------
__global__ __launch_bounds__(256) void softmax_kernel(const float* __restrict__ scores,
                                                      float* __restrict__ attn) {
  __shared__ float red[256];
  const int b = blockIdx.x, tid = threadIdx.x;
  const float* row = scores + (size_t)b * S_;
  float v[8];
  float m = -INFINITY;
#pragma unroll
  for (int i = 0; i < 8; ++i) { v[i] = row[tid + i * 256]; m = fmaxf(m, v[i]); }
  red[tid] = m; __syncthreads();
  for (int st = 128; st > 0; st >>= 1) {
    if (tid < st) red[tid] = fmaxf(red[tid], red[tid + st]);
    __syncthreads();
  }
  m = red[0]; __syncthreads();
  float sum = 0.0f;
#pragma unroll
  for (int i = 0; i < 8; ++i) { v[i] = __expf(v[i] - m); sum += v[i]; }
  red[tid] = sum; __syncthreads();
  for (int st = 128; st > 0; st >>= 1) {
    if (tid < st) red[tid] += red[tid + st];
    __syncthreads();
  }
  float inv = 1.0f / red[0];
#pragma unroll
  for (int i = 0; i < 8; ++i) attn[(size_t)b * S_ + tid + i * 256] = v[i] * inv;
}

// ---------------- Pass 3: zero output (d_out is poisoned) ------------------
__global__ __launch_bounds__(256) void zero_kernel(float* __restrict__ out) {
  out[blockIdx.x * 256 + threadIdx.x] = 0.0f;
}

// ---------------- Pass 4: out[b,d] = sum_s attn[b,s]*enc[s,b,d] ------------
// grid (D/256, B, 16): each block sums a 128-long s-slice, f32 atomics.
__global__ __launch_bounds__(256) void wsum_kernel(const float* __restrict__ enc,
                                                   const float* __restrict__ attn,
                                                   float* __restrict__ out) {
  __shared__ float aw[128];
  const int d  = blockIdx.x * 256 + threadIdx.x;
  const int b  = blockIdx.y;
  const int s0 = blockIdx.z * 128;
  if (threadIdx.x < 128) aw[threadIdx.x] = attn[(size_t)b * S_ + s0 + threadIdx.x];
  __syncthreads();
  float acc = 0.0f;
#pragma unroll 4
  for (int j = 0; j < 128; ++j)
    acc = fmaf(aw[j], enc[((size_t)(s0 + j) * B_ + b) * D_ + d], acc);
  atomicAdd(out + (size_t)b * D_ + d, acc);
}

extern "C" void kernel_launch(void* const* d_in, const int* in_sizes, int n_in,
                              void* d_out, int out_size, void* d_ws, size_t ws_size,
                              hipStream_t stream) {
  const float* enc = (const float*)d_in[0];   // (S,B,D)
  const float* dec = (const float*)d_in[1];   // (B,D)
  const float* Wfc = (const float*)d_in[2];   // (E,D)
  const float* sw  = (const float*)d_in[3];   // (E,1)
  float* out = (float*)d_out;                 // (B,D)

  char* ws = (char*)d_ws;
  __bf16* Wb     = (__bf16*)ws;                              // 2 MB
  float*  scores = (float*)(ws + (2u << 20));                // 256 KB (B,S)
  float*  attn   = (float*)(ws + (2u << 20) + (256u << 10)); // 256 KB (B,S)

  wconv_kernel  <<<(E_ * D_) / 1024, 256, 0, stream>>>(Wfc, Wb);
  score_kernel  <<<S_, 512, 0, stream>>>(enc, dec, Wb, sw, scores);
  softmax_kernel<<<B_, 256, 0, stream>>>(scores, attn);
  zero_kernel   <<<(B_ * D_) / 256, 256, 0, stream>>>(out);
  wsum_kernel   <<<dim3(D_ / 256, B_, 16), 256, 0, stream>>>(enc, attn, out);
}
